// GCRNTransformer_43258910605765
// MI455X (gfx1250) — matrix operand order repaired and checked by reading, not compiled, not run on hardware
//
#include <hip/hip_runtime.h>
#include <hip/hip_bf16.h>

typedef _Float16 f16;
typedef __attribute__((ext_vector_type(16))) _Float16 v16h;
typedef __attribute__((ext_vector_type(8)))  _Float16 v8h;
typedef __attribute__((ext_vector_type(4)))  _Float16 v4h;
typedef __attribute__((ext_vector_type(8)))  float    v8f;

#define TT 12

// LDS leading dimensions (16B-aligned rows + bank-skewed)
constexpr int LDA  = 72;   // f16 16-row activation tile (halves); row = 144B, skew 36 dwords
constexpr int LDH  = 280;  // f16 16x256 hidden tile (halves);    row = 560B, skew 12 dwords
constexpr int LDXF = 68;   // f32 16-row tile (floats);           row = 272B, skew 4 banks

// swizzled weights: subtile (kt,nt) of 32x16 stored as [lane][16 halves]
constexpr int W64_SWZ = 64 * 64;    // halves
constexpr int W1_SWZ  = 64 * 256;
constexpr int W2_SWZ  = 256 * 64;

constexpr size_t TWB    = (size_t)(4 * W64_SWZ + W1_SWZ + W2_SWZ) * 2;  // 98304 B
constexpr int    PWT    = 4352 + 2304 + 2304 + 8960;                    // 17920 B per wave
constexpr size_t SMEM_T = TWB + 8 * (size_t)PWT;                        // ~236 KB

constexpr int    PWG     = 2304 + 2304 + 64 + 64;                 // 4736 B per wave
constexpr size_t SMEM_ZR = (size_t)4 * W64_SWZ * 2 + 8 * PWG;     // 70656 B
constexpr size_t SMEM_G  = (size_t)2 * W64_SWZ * 2 + 8 * PWG;     // 54272 B

// ---------------------------------------------------------------- helpers

__device__ __forceinline__ void wsync() {
  // wave-local LDS ordering (wave32 lockstep; stops compiler reordering too)
  asm volatile("s_wait_dscnt 0x0" ::: "memory");
}

__device__ __forceinline__ v8f wmma16(v16h a, v16h b, v8f c) {
  return __builtin_amdgcn_wmma_f32_16x16x32_f16(
      /*neg_a=*/false, a, /*neg_b=*/false, b,
      /*c_mod=*/(short)0, c, /*reuse_a=*/false, /*reuse_b=*/false);
}

// A fragment: 16x32 f16; lane row = lane&15; two aligned v8h group loads
// (K = hg*8 + 0..7 at p, K = 16 + hg*8 + 0..7 at p+16) per ISA 7.12.2.
__device__ __forceinline__ v16h load_a_frag(const f16* A, int lda, int lane) {
  const int row = lane & 15;
  const int hg  = lane >> 4;
  const f16* p = A + row * lda + hg * 8;
  v8h lo = *(const v8h*)p;
  v8h hi = *(const v8h*)(p + 16);
  return __builtin_shufflevector(lo, hi, 0, 1, 2, 3, 4, 5, 6, 7,
                                 8, 9, 10, 11, 12, 13, 14, 15);
}

// swizzled weight staging: element (k,n) -> subtile (kt = k/32, nt = n/16),
// lane = ((k>>3)&1)*16 + (n&15), element i = (k&7) + ((k>>4)&1)*8; each lane's 16 halves of a
// B fragment are contiguous (32 B) -> 2x ds_load_b128 at use.
__device__ __forceinline__ void stage_swz(f16* dst, const float* __restrict__ src,
                                          int K, int NC, int tid, int nth) {
  const int ntiles = NC >> 4;
  for (int idx = tid; idx < K * NC; idx += nth) {
    const int k = idx / NC, n = idx - k * NC;
    const int kt = k >> 5;
    const int lane = ((k >> 3) & 1) * 16 + (n & 15);
    const int i = (k & 7) + ((k >> 4) & 1) * 8;
    const int nt = n >> 4;
    dst[(((kt * ntiles + nt) * 32) + lane) * 16 + i] = (f16)src[idx];
  }
}

__device__ __forceinline__ v16h load_b_swz(const f16* Wswz, int ntiles, int kt,
                                           int nt, int lane) {
  return *(const v16h*)(Wswz + (((kt * ntiles + nt) * 32) + lane) * 16);
}

__device__ __forceinline__ float sigm(float x) { return 1.f / (1.f + __expf(-x)); }
__device__ __forceinline__ float tanh_c(float x) {
  x = fminf(fmaxf(x, -15.f), 15.f);
  float e = __expf(2.f * x);
  return (e - 1.f) / (e + 1.f);
}

// per-(head, query-row) attention over the 12x12 window; vectorized v8h row reads
__device__ __forceinline__ void attn_pair(const f16* qh, const f16* kh,
                                          const f16* vh, f16* ahh, int p) {
  const int head = p / 12;
  const int tq = p - head * 12;
  const int cb = head * 16;
  const v8h* qp = (const v8h*)(qh + tq * LDA + cb);
  const v8h q0 = qp[0], q1 = qp[1];
  float s[12];
  float mxs = -1e30f;
#pragma unroll
  for (int tk = 0; tk < 12; ++tk) {
    const v8h* kp = (const v8h*)(kh + tk * LDA + cb);
    const v8h k0 = kp[0], k1 = kp[1];
    float acc = 0.f;
#pragma unroll
    for (int d = 0; d < 8; ++d)
      acc += (float)q0[d] * (float)k0[d] + (float)q1[d] * (float)k1[d];
    s[tk] = acc * 0.25f;  // 1/sqrt(16)
    mxs = fmaxf(mxs, s[tk]);
  }
  float sum = 0.f;
#pragma unroll
  for (int tk = 0; tk < 12; ++tk) { s[tk] = __expf(s[tk] - mxs); sum += s[tk]; }
  const float inv = 1.f / sum;
  float o[16];
#pragma unroll
  for (int d = 0; d < 16; ++d) o[d] = 0.f;
#pragma unroll
  for (int tk = 0; tk < 12; ++tk) {
    const v8h* vp = (const v8h*)(vh + tk * LDA + cb);
    const v8h va = vp[0], vb = vp[1];
    const float w = s[tk];
#pragma unroll
    for (int d = 0; d < 8; ++d) { o[d] += w * (float)va[d]; o[8 + d] += w * (float)vb[d]; }
  }
  v8h r0, r1;
#pragma unroll
  for (int d = 0; d < 8; ++d) {
    r0[d] = (f16)(o[d] * inv);
    r1[d] = (f16)(o[8 + d] * inv);
  }
  v8h* op = (v8h*)(ahh + tq * LDA + cb);
  op[0] = r0;
  op[1] = r1;
}

// ---------------------------------------------------------------- kernels

__global__ void __launch_bounds__(256) zero_kernel(float* __restrict__ p, int n) {
  int i = blockIdx.x * 256 + threadIdx.x;
  if (i < n) p[i] = 0.f;
}

// one wave = one node; 8 waves/block; persistent over node blocks.
__global__ void __launch_bounds__(256) transformer_kernel(
    const float* __restrict__ x_seq,
    const float* __restrict__ Wpi, const float* __restrict__ bpi,
    const float* __restrict__ Wq,  const float* __restrict__ Wk,
    const float* __restrict__ Wv,  const float* __restrict__ Wo,
    const float* __restrict__ W1,  const float* __restrict__ W2,
    const float* __restrict__ Wpo,
    const float* __restrict__ bq,  const float* __restrict__ bk,
    const float* __restrict__ bv,  const float* __restrict__ bo,
    const float* __restrict__ b1v, const float* __restrict__ b2v,
    const float* __restrict__ bpo,
    const float* __restrict__ g1,  const float* __restrict__ be1,
    const float* __restrict__ g2,  const float* __restrict__ be2,
    float* __restrict__ x2, int N)
{
  extern __shared__ char smem[];
  f16* wqS = (f16*)smem;
  f16* wkS = wqS + W64_SWZ;
  f16* wvS = wkS + W64_SWZ;
  f16* woS = wvS + W64_SWZ;
  f16* w1S = woS + W64_SWZ;
  f16* w2S = w1S + W1_SWZ;

  const int tid = threadIdx.x;
  stage_swz(wqS, Wq, 64, 64, tid, 256);
  stage_swz(wkS, Wk, 64, 64, tid, 256);
  stage_swz(wvS, Wv, 64, 64, tid, 256);
  stage_swz(woS, Wo, 64, 64, tid, 256);
  stage_swz(w1S, W1, 64, 256, tid, 256);
  stage_swz(w2S, W2, 256, 64, tid, 256);
  __syncthreads();

  const int wave = tid >> 5;
  const int lane = tid & 31;
  const int hg = lane >> 4, nn = lane & 15;

  char* pw = smem + TWB + (size_t)wave * PWT;
  float* xf  = (float*)pw;               // f32 [16][68]
  f16*   xh  = (f16*)(pw + 4352);        // f16 [16][72]
  f16*   ahh = (f16*)(pw + 6656);        // f16 [16][72]
  f16*   un  = (f16*)(pw + 8960);        // union: qkv | hidden
  f16*   qh  = un;
  f16*   kh  = un + 16 * LDA;
  f16*   vh  = un + 32 * LDA;
  f16*   hid = un;                       // f16 [16][280]

  for (int base = blockIdx.x * 8; base < N; base += gridDim.x * 8) {
    const int node = base + wave;
    if (node >= N) continue;  // wave-uniform; no block barriers inside

    // 1: input projection x[t][d] = xs[t]*Wpi[d] + bpi[d]; zero-pad rows 12..15
    for (int idx = lane; idx < 16 * 16; idx += 32) {
      const int r = idx >> 4, c4 = (idx & 15) << 2;
      float4 o;
      if (r < TT) {
        const float4 wv = *(const float4*)(Wpi + c4);
        const float4 bv = *(const float4*)(bpi + c4);
        const float xs = x_seq[r * N + node];
        o = make_float4(xs * wv.x + bv.x, xs * wv.y + bv.y,
                        xs * wv.z + bv.z, xs * wv.w + bv.w);
      } else {
        o = make_float4(0.f, 0.f, 0.f, 0.f);
      }
      *(float4*)(xf + r * LDXF + c4) = o;
      v4h h4;
      h4[0] = (f16)o.x; h4[1] = (f16)o.y; h4[2] = (f16)o.z; h4[3] = (f16)o.w;
      *(v4h*)(xh + r * LDA + c4) = h4;
    }
    wsync();

    // 2: q,k,v GEMMs (16x64 @ 64x64)
    {
      v16h a0 = load_a_frag(xh, LDA, lane);
      v16h a1 = load_a_frag(xh + 32, LDA, lane);
#pragma unroll
      for (int nt = 0; nt < 4; ++nt) {
        const int col = nt * 16 + nn;
        v8f cq = {}, ck = {}, cv = {};
        cq = wmma16(a0, load_b_swz(wqS, 4, 0, nt, lane), cq);
        cq = wmma16(a1, load_b_swz(wqS, 4, 1, nt, lane), cq);
        ck = wmma16(a0, load_b_swz(wkS, 4, 0, nt, lane), ck);
        ck = wmma16(a1, load_b_swz(wkS, 4, 1, nt, lane), ck);
        cv = wmma16(a0, load_b_swz(wvS, 4, 0, nt, lane), cv);
        cv = wmma16(a1, load_b_swz(wvS, 4, 1, nt, lane), cv);
        const float bqc = bq[col], bkc = bk[col], bvc = bv[col];
#pragma unroll
        for (int v = 0; v < 8; ++v) {
          const int row = v + hg * 8;
          qh[row * LDA + col] = (f16)(cq[v] + bqc);
          kh[row * LDA + col] = (f16)(ck[v] + bkc);
          vh[row * LDA + col] = (f16)(cv[v] + bvc);
        }
      }
    }
    wsync();

    // 3: attention (48 (head,row) pairs over 32 lanes)
    attn_pair(qh, kh, vh, ahh, lane);
    if (lane < 16) attn_pair(qh, kh, vh, ahh, lane + 32);
    {  // zero-pad rows 12..15 of ahh (4 rows x 64 halves = 32 v8h chunks)
      const int r = 12 + (lane >> 3), c8 = (lane & 7) << 3;
      v8h zz = {};
      *(v8h*)(ahh + r * LDA + c8) = zz;
    }
    wsync();

    // 4: output proj Wo + residual into xf
    {
      v16h a0 = load_a_frag(ahh, LDA, lane);
      v16h a1 = load_a_frag(ahh + 32, LDA, lane);
#pragma unroll
      for (int nt = 0; nt < 4; ++nt) {
        const int col = nt * 16 + nn;
        v8f cc = {};
        cc = wmma16(a0, load_b_swz(woS, 4, 0, nt, lane), cc);
        cc = wmma16(a1, load_b_swz(woS, 4, 1, nt, lane), cc);
        const float boc = bo[col];
#pragma unroll
        for (int v = 0; v < 8; ++v)
          xf[(v + hg * 8) * LDXF + col] += cc[v] + boc;
      }
    }
    wsync();

    // 5: LayerNorm 1 (row per lane)
    if (lane < TT) {
      float* row = xf + lane * LDXF;
      float m = 0.f;
#pragma unroll
      for (int c = 0; c < 64; ++c) m += row[c];
      m *= (1.f / 64.f);
      float var = 0.f;
#pragma unroll
      for (int c = 0; c < 64; ++c) { float d = row[c] - m; var += d * d; }
      var *= (1.f / 64.f);
      const float inv = rsqrtf(var + 1e-5f);
#pragma unroll
      for (int c = 0; c < 64; ++c) {
        float v = (row[c] - m) * inv * g1[c] + be1[c];
        row[c] = v;
        xh[lane * LDA + c] = (f16)v;
      }
    }
    wsync();

    // 6: FFN up 64->256 + ReLU
    {
      v16h a0 = load_a_frag(xh, LDA, lane);
      v16h a1 = load_a_frag(xh + 32, LDA, lane);
#pragma unroll
      for (int nt = 0; nt < 16; ++nt) {
        const int col = nt * 16 + nn;
        v8f cc = {};
        cc = wmma16(a0, load_b_swz(w1S, 16, 0, nt, lane), cc);
        cc = wmma16(a1, load_b_swz(w1S, 16, 1, nt, lane), cc);
        const float bc = b1v[col];
#pragma unroll
        for (int v = 0; v < 8; ++v)
          hid[(v + hg * 8) * LDH + col] = (f16)fmaxf(cc[v] + bc, 0.f);
      }
    }
    wsync();

    // 7: FFN down 256->64 + residual
    {
      v16h af[8];
#pragma unroll
      for (int kt = 0; kt < 8; ++kt) af[kt] = load_a_frag(hid + kt * 32, LDH, lane);
#pragma unroll
      for (int nt = 0; nt < 4; ++nt) {
        const int col = nt * 16 + nn;
        v8f cc = {};
#pragma unroll
        for (int kt = 0; kt < 8; ++kt)
          cc = wmma16(af[kt], load_b_swz(w2S, 4, kt, nt, lane), cc);
        const float bc = b2v[col];
#pragma unroll
        for (int v = 0; v < 8; ++v)
          xf[(v + hg * 8) * LDXF + col] += cc[v] + bc;
      }
    }
    wsync();

    // 8: LayerNorm 2 fused with Wpo projection (64 -> 1)
    if (lane < TT) {
      const float* row = xf + lane * LDXF;
      float m = 0.f;
#pragma unroll
      for (int c = 0; c < 64; ++c) m += row[c];
      m *= (1.f / 64.f);
      float var = 0.f;
#pragma unroll
      for (int c = 0; c < 64; ++c) { float d = row[c] - m; var += d * d; }
      var *= (1.f / 64.f);
      const float inv = rsqrtf(var + 1e-5f);
      float acc = 0.f;
#pragma unroll
      for (int c = 0; c < 64; ++c)
        acc += ((row[c] - m) * inv * g2[c] + be2[c]) * Wpo[c];
      x2[lane * N + node] = acc + bpo[0];  // xs layout [t][node]
    }
    wsync();
  }
}

// scalar scatter: mx[dst] += ew * xt[src]   (F = 1)
__global__ void __launch_bounds__(256) scatter_scalar_kernel(
    const int* __restrict__ src, const int* __restrict__ dst,
    const float* __restrict__ ew, const float* __restrict__ x,
    float* __restrict__ mx, int E)
{
  int e = blockIdx.x * 256 + threadIdx.x;
  if (e < E) atomicAdd(&mx[dst[e]], ew[e] * x[src[e]]);
}

// 64-dim scatter: msg[dst][:] += ew * y[src][:]  (float4 per thread)
__global__ void __launch_bounds__(256) scatter_vec_kernel(
    const int* __restrict__ src, const int* __restrict__ dst,
    const float* __restrict__ ew, const float* __restrict__ y,
    float* __restrict__ msg, int E)
{
  int i = blockIdx.x * 256 + threadIdx.x;
  if (i < E * 16) {
    const int e = i >> 4, c = (i & 15) << 2;
    const float w = ew[e];
    const float4 vv = *(const float4*)(y + (size_t)src[e] * 64 + c);
    float* m = msg + (size_t)dst[e] * 64 + c;
    atomicAdd(m + 0, w * vv.x);
    atomicAdd(m + 1, w * vv.y);
    atomicAdd(m + 2, w * vv.z);
    atomicAdd(m + 3, w * vv.w);
  }
}

// load a 16-node x 64 f32 tile -> f16 LDS tile (float4 + packed v4h stores)
__device__ __forceinline__ void load_tile_f16(f16* dstA, f16* dstB,
                                              const float* __restrict__ srcA,
                                              const float* __restrict__ srcB,
                                              int base, int N, int lane) {
  for (int idx = lane; idx < 16 * 16; idx += 32) {
    const int r = idx >> 4, c4 = (idx & 15) << 2;
    const int n = base + r;
    v4h a = {}, b = {};
    if (n < N) {
      const float4 va = *(const float4*)(srcA + (size_t)n * 64 + c4);
      const float4 vb = *(const float4*)(srcB + (size_t)n * 64 + c4);
      a[0] = (f16)va.x; a[1] = (f16)va.y; a[2] = (f16)va.z; a[3] = (f16)va.w;
      b[0] = (f16)vb.x; b[1] = (f16)vb.y; b[2] = (f16)vb.z; b[3] = (f16)vb.w;
    }
    *(v4h*)(dstA + r * LDA + c4) = a;
    *(v4h*)(dstB + r * LDA + c4) = b;
  }
}

// z,r gates: z = sigmoid(h@Cz + msgH@Dz + cz + xt*Az + mx*Bz + bz), r likewise;
// also s = r * h for the candidate-state scatter. One 16-node tile per wave.
__global__ void __launch_bounds__(256) gru_zr_kernel(
    const float* __restrict__ xt,  const float* __restrict__ mxv,
    const float* __restrict__ hst, const float* __restrict__ msgH,
    const float* __restrict__ Cz, const float* __restrict__ Dz, const float* __restrict__ cz,
    const float* __restrict__ Az, const float* __restrict__ Bz, const float* __restrict__ bz,
    const float* __restrict__ Cr, const float* __restrict__ Dr, const float* __restrict__ cr,
    const float* __restrict__ Ar, const float* __restrict__ Br, const float* __restrict__ br,
    float* __restrict__ zbuf, float* __restrict__ sbuf, int N)
{
  extern __shared__ char smem[];
  f16* czS = (f16*)smem;
  f16* dzS = czS + W64_SWZ;
  f16* crS = dzS + W64_SWZ;
  f16* drS = crS + W64_SWZ;
  const int tid = threadIdx.x;
  stage_swz(czS, Cz, 64, 64, tid, 256);
  stage_swz(dzS, Dz, 64, 64, tid, 256);
  stage_swz(crS, Cr, 64, 64, tid, 256);
  stage_swz(drS, Dr, 64, 64, tid, 256);
  __syncthreads();
  const int wave = tid >> 5, lane = tid & 31;
  const int hg = lane >> 4, nn = lane & 15;
  char* pw = smem + (size_t)4 * W64_SWZ * 2 + (size_t)wave * PWG;
  f16*   hh  = (f16*)pw;
  f16*   mh  = (f16*)(pw + 2304);
  float* xtl = (float*)(pw + 4608);
  float* mxl = (float*)(pw + 4672);
  const int base = blockIdx.x * 128 + wave * 16;
  if (base >= N) return;

  load_tile_f16(hh, mh, hst, msgH, base, N, lane);
  if (lane < 16) {
    const int n = base + lane;
    xtl[lane] = (n < N) ? xt[n] : 0.f;
    mxl[lane] = (n < N) ? mxv[n] : 0.f;
  }
  wsync();

  v16h h0 = load_a_frag(hh, LDA, lane);
  v16h h1 = load_a_frag(hh + 32, LDA, lane);
  v16h m0 = load_a_frag(mh, LDA, lane);
  v16h m1 = load_a_frag(mh + 32, LDA, lane);
#pragma unroll
  for (int nt = 0; nt < 4; ++nt) {
    const int col = nt * 16 + nn;
    v8f az = {}, ar = {};
    az = wmma16(h0, load_b_swz(czS, 4, 0, nt, lane), az);
    az = wmma16(h1, load_b_swz(czS, 4, 1, nt, lane), az);
    az = wmma16(m0, load_b_swz(dzS, 4, 0, nt, lane), az);
    az = wmma16(m1, load_b_swz(dzS, 4, 1, nt, lane), az);
    ar = wmma16(h0, load_b_swz(crS, 4, 0, nt, lane), ar);
    ar = wmma16(h1, load_b_swz(crS, 4, 1, nt, lane), ar);
    ar = wmma16(m0, load_b_swz(drS, 4, 0, nt, lane), ar);
    ar = wmma16(m1, load_b_swz(drS, 4, 1, nt, lane), ar);
    const float azc = Az[col], bzc = Bz[col], czc = bz[col] + cz[col];
    const float arc = Ar[col], brc = Br[col], crc = br[col] + cr[col];
#pragma unroll
    for (int v = 0; v < 8; ++v) {
      const int row = v + hg * 8;
      const int n = base + row;
      if (n < N) {
        const float xv = xtl[row], mv = mxl[row];
        const float zg = sigm(az[v] + xv * azc + mv * bzc + czc);
        const float rg = sigm(ar[v] + xv * arc + mv * brc + crc);
        zbuf[(size_t)n * 64 + col] = zg;
        sbuf[(size_t)n * 64 + col] = rg * hst[(size_t)n * 64 + col];
      }
    }
  }
}

// g = tanh(s@Ch + msgS@Dh + cg + xt*Ah + mx*Bh + bg); h = z*h + (1-z)*g
__global__ void __launch_bounds__(256) gru_g_kernel(
    const float* __restrict__ xt,   const float* __restrict__ mxv,
    const float* __restrict__ sbuf, const float* __restrict__ msgS,
    const float* __restrict__ Ch, const float* __restrict__ Dh, const float* __restrict__ cg,
    const float* __restrict__ Ah, const float* __restrict__ Bh, const float* __restrict__ bg,
    const float* __restrict__ zbuf, float* __restrict__ hbuf, int N)
{
  extern __shared__ char smem[];
  f16* chS = (f16*)smem;
  f16* dhS = chS + W64_SWZ;
  const int tid = threadIdx.x;
  stage_swz(chS, Ch, 64, 64, tid, 256);
  stage_swz(dhS, Dh, 64, 64, tid, 256);
  __syncthreads();
  const int wave = tid >> 5, lane = tid & 31;
  const int hg = lane >> 4, nn = lane & 15;
  char* pw = smem + (size_t)2 * W64_SWZ * 2 + (size_t)wave * PWG;
  f16*   sh  = (f16*)pw;
  f16*   mh  = (f16*)(pw + 2304);
  float* xtl = (float*)(pw + 4608);
  float* mxl = (float*)(pw + 4672);
  const int base = blockIdx.x * 128 + wave * 16;
  if (base >= N) return;

  load_tile_f16(sh, mh, sbuf, msgS, base, N, lane);
  if (lane < 16) {
    const int n = base + lane;
    xtl[lane] = (n < N) ? xt[n] : 0.f;
    mxl[lane] = (n < N) ? mxv[n] : 0.f;
  }
  wsync();

  v16h s0 = load_a_frag(sh, LDA, lane);
  v16h s1 = load_a_frag(sh + 32, LDA, lane);
  v16h m0 = load_a_frag(mh, LDA, lane);
  v16h m1 = load_a_frag(mh + 32, LDA, lane);
#pragma unroll
  for (int nt = 0; nt < 4; ++nt) {
    const int col = nt * 16 + nn;
    v8f ag = {};
    ag = wmma16(s0, load_b_swz(chS, 4, 0, nt, lane), ag);
    ag = wmma16(s1, load_b_swz(chS, 4, 1, nt, lane), ag);
    ag = wmma16(m0, load_b_swz(dhS, 4, 0, nt, lane), ag);
    ag = wmma16(m1, load_b_swz(dhS, 4, 1, nt, lane), ag);
    const float ahc = Ah[col], bhc = Bh[col], cgc = bg[col] + cg[col];
#pragma unroll
    for (int v = 0; v < 8; ++v) {
      const int row = v + hg * 8;
      const int n = base + row;
      if (n < N) {
        const float gg = tanh_c(ag[v] + xtl[row] * ahc + mxl[row] * bhc + cgc);
        const float zg = zbuf[(size_t)n * 64 + col];
        const float hv = hbuf[(size_t)n * 64 + col];
        hbuf[(size_t)n * 64 + col] = zg * hv + (1.f - zg) * gg;
      }
    }
  }
}

// out[n] = relu(h[n]) . Whd + bhd
__global__ void __launch_bounds__(256) head_kernel(
    const float* __restrict__ h, const float* __restrict__ Whd,
    const float* __restrict__ bhd, float* __restrict__ out, int N)
{
  int n = blockIdx.x * 256 + threadIdx.x;
  if (n < N) {
    float acc = 0.f;
#pragma unroll
    for (int k = 0; k < 64; ++k)
      acc += fmaxf(h[(size_t)n * 64 + k], 0.f) * Whd[k];
    out[n] = acc + bhd[0];
  }
}

// ---------------------------------------------------------------- launch

extern "C" void kernel_launch(void* const* d_in, const int* in_sizes, int n_in,
                              void* d_out, int out_size, void* d_ws, size_t ws_size,
                              hipStream_t stream)
{
  (void)n_in; (void)out_size; (void)ws_size;
  const float* x_seq = (const float*)d_in[0];
  const int*   ei    = (const int*)d_in[1];
  const float* ew    = (const float*)d_in[2];
  const float* Wpi = (const float*)d_in[3];
  const float* bpi = (const float*)d_in[4];
  const float* Wq  = (const float*)d_in[5];
  const float* Wk  = (const float*)d_in[6];
  const float* Wv  = (const float*)d_in[7];
  const float* Wo  = (const float*)d_in[8];
  const float* W1  = (const float*)d_in[9];
  const float* W2  = (const float*)d_in[10];
  const float* Wpo = (const float*)d_in[11];
  const float* bq  = (const float*)d_in[12];
  const float* bk  = (const float*)d_in[13];
  const float* bv  = (const float*)d_in[14];
  const float* bo  = (const float*)d_in[15];
  const float* b1  = (const float*)d_in[16];
  const float* b2  = (const float*)d_in[17];
  const float* bpo = (const float*)d_in[18];
  const float* g1  = (const float*)d_in[19];
  const float* be1 = (const float*)d_in[20];
  const float* g2  = (const float*)d_in[21];
  const float* be2 = (const float*)d_in[22];
  const float* Az = (const float*)d_in[23];
  const float* Bz = (const float*)d_in[24];
  const float* bz = (const float*)d_in[25];
  const float* Cz = (const float*)d_in[26];
  const float* Dz = (const float*)d_in[27];
  const float* cz = (const float*)d_in[28];
  const float* Ar = (const float*)d_in[29];
  const float* Br = (const float*)d_in[30];
  const float* br = (const float*)d_in[31];
  const float* Cr = (const float*)d_in[32];
  const float* Dr = (const float*)d_in[33];
  const float* cr = (const float*)d_in[34];
  const float* Ah = (const float*)d_in[35];
  const float* Bh = (const float*)d_in[36];
  const float* bg = (const float*)d_in[37];
  const float* Ch = (const float*)d_in[38];
  const float* Dh = (const float*)d_in[39];
  const float* cg = (const float*)d_in[40];
  const float* Whd = (const float*)d_in[41];
  const float* bhd = (const float*)d_in[42];
  float* out = (float*)d_out;

  const int N = in_sizes[0] / TT;  // B=F=1, x_seq = B*T*N*F
  const int E = in_sizes[2];
  const int* src = ei;
  const int* dst = ei + E;

  float* ws   = (float*)d_ws;
  float* x2   = ws;                        // [T][N]
  float* hbuf = x2 + (size_t)TT * N;       // [N][64]
  float* sbuf = hbuf + (size_t)N * 64;     // [N][64]
  float* zbuf = sbuf + (size_t)N * 64;     // [N][64]
  float* msgH = zbuf + (size_t)N * 64;     // [N][64]
  float* msgS = msgH + (size_t)N * 64;     // [N][64]   (contiguous with msgH)
  float* mx   = msgS + (size_t)N * 64;     // [N]       (contiguous with msgS)

  // h0 = 0 (deterministic re-init every call)
  {
    int cnt = N * 64;
    zero_kernel<<<(cnt + 255) / 256, 256, 0, stream>>>(hbuf, cnt);
  }

  transformer_kernel<<<640, 256, SMEM_T, stream>>>(
      x_seq, Wpi, bpi, Wq, Wk, Wv, Wo, W1, W2, Wpo,
      bq, bk, bv, bo, b1, b2, bpo, g1, be1, g2, be2, x2, N);

  const int tiles = (N + 127) / 128;
  for (int t = 0; t < TT; ++t) {
    const float* xt = x2 + (size_t)t * N;
    const int zc = N * 64 * 2 + N;  // msgH + msgS + mx (contiguous)
    zero_kernel<<<(zc + 255) / 256, 256, 0, stream>>>(msgH, zc);
    scatter_scalar_kernel<<<(E + 255) / 256, 256, 0, stream>>>(src, dst, ew, xt, mx, E);
    scatter_vec_kernel<<<(E * 16 + 255) / 256, 256, 0, stream>>>(src, dst, ew, hbuf, msgH, E);
    gru_zr_kernel<<<tiles, 256, SMEM_ZR, stream>>>(
        xt, mx, hbuf, msgH, Cz, Dz, cz, Az, Bz, bz, Cr, Dr, cr, Ar, Br, br,
        zbuf, sbuf, N);
    scatter_vec_kernel<<<(E * 16 + 255) / 256, 256, 0, stream>>>(src, dst, ew, sbuf, msgS, E);
    gru_g_kernel<<<tiles, 256, SMEM_G, stream>>>(
        xt, mx, sbuf, msgS, Ch, Dh, cg, Ah, Bh, bg, zbuf, hbuf, N);
  }

  head_kernel<<<(N + 255) / 256, 256, 0, stream>>>(hbuf, Whd, bhd, out, N);
}
